// GCN_corr_class_v2_23158463660684
// MI455X (gfx1250) — compile-verified, matrix operand order checked by reading, not run on hardware
//
#include <hip/hip_runtime.h>
#include <hip/hip_bf16.h>

// ---------------------------------------------------------------------------
// GCN_corr_class_v2 fully-fused CDNA5 (gfx1250) implementation.
//   - all GEMMs via v_wmma_f32_16x16x32_f16 (f32 accumulate)
//   - per-sample fused gconv: adj @ (X @ W), intermediate lives in LDS
//   - gconv bias skipped wherever BatchNorm follows (cancels exactly)
//   - X-tile staging via TDM tensor_load_to_lds (TENSORcnt) when available;
//     adj staging via GLOBAL_LOAD_ASYNC_TO_LDS_B128 (ASYNCcnt);
//     falls back to plain b128 copies when builtins are missing
// ---------------------------------------------------------------------------

#define NB     8192
#define NN     57
#define INF    25
#define HID    128
#define CLS    12
#define BN_EPS 1e-5f

typedef _Float16 h16;
typedef __attribute__((ext_vector_type(16))) _Float16 v16h;
typedef __attribute__((ext_vector_type(8)))  float    v8f;
typedef __attribute__((ext_vector_type(4)))  int      v4i;
typedef __attribute__((ext_vector_type(4)))  unsigned u32x4;
typedef __attribute__((ext_vector_type(8)))  int      i32x8;
typedef __attribute__((ext_vector_type(4)))  int      i32x4;

#if defined(__has_builtin)
#if __has_builtin(__builtin_amdgcn_global_load_async_to_lds_b128)
#define USE_ASYNC_LDS 1
#endif
#if __has_builtin(__builtin_amdgcn_tensor_load_to_lds)
#define USE_TDM 1
#if defined(__has_include)
#if __has_include(<hip/amd_detail/amd_gfx1250_TDM.h>)
#define TDM_SIX_ARGS 1
#endif
#endif
#endif
#endif

#ifdef USE_ASYNC_LDS
typedef __attribute__((address_space(1))) v4i gv4i;  // global-AS int4 vector
typedef __attribute__((address_space(3))) v4i lv4i;  // LDS-AS int4 vector
__device__ inline void async_wait0() {
#if __has_builtin(__builtin_amdgcn_s_wait_asynccnt)
  __builtin_amdgcn_s_wait_asynccnt(0);
#else
  asm volatile("s_wait_asynccnt 0x0" ::: "memory");
#endif
}
#endif

#ifdef USE_TDM
__device__ inline void tensor_wait0() {
#if __has_builtin(__builtin_amdgcn_s_wait_tensorcnt)
  __builtin_amdgcn_s_wait_tensorcnt(0);
#else
  asm volatile("s_wait_tensorcnt 0x0" ::: "memory");
#endif
}

// One-shot 2D tile DMA: global [rows][cols] f16 (row-major, stride=cols)
// -> LDS at ldst.  D# per CDNA5 ISA (groups 2/3 zero).
__device__ inline void tdm_load_2d(h16* ldst, const h16* gsrc, int rows,
                                   int cols) {
  unsigned lds_off =
      (unsigned)(size_t)(__attribute__((address_space(3))) h16*)ldst;
  unsigned long long ga = (unsigned long long)(size_t)gsrc;
  u32x4 g0;
  g0[0] = 1u;                                           // count=1 (valid D#)
  g0[1] = lds_off;                                      // lds_addr [63:32]
  g0[2] = (unsigned)ga;                                 // global_addr[31:0]
  g0[3] = (unsigned)((ga >> 32) & 0x1FFFFFFull) | (2u << 30);  // [56:32]+type=2
  const unsigned td0 = (unsigned)cols, td1 = (unsigned)rows;
  const unsigned long long st0 = (unsigned long long)cols;  // dim0 stride
  i32x8 g1;
  g1[0] = (int)(1u << 16);                              // data_size=1 -> 2B
  g1[1] = (int)((td0 & 0xFFFFu) << 16);                 // tensor_dim0 lo16
  g1[2] = (int)((td0 >> 16) | ((td1 & 0xFFFFu) << 16)); // dim0 hi / dim1 lo
  g1[3] = (int)((td1 >> 16) | ((unsigned)cols << 16));  // dim1 hi / tile_dim0
  g1[4] = (int)(unsigned)rows;                          // tile_dim1 (tile2=0)
  g1[5] = (int)(unsigned)(st0 & 0xFFFFFFFFull);         // dim0_stride[31:0]
  g1[6] = (int)(unsigned)((st0 >> 32) & 0xFFFFull);     // stride hi / d1s lo
  g1[7] = 0;
  i32x4 z4 = {0, 0, 0, 0};
#ifdef TDM_SIX_ARGS
  i32x8 z8 = {0, 0, 0, 0, 0, 0, 0, 0};
  __builtin_amdgcn_tensor_load_to_lds(g0, g1, z4, z4, z8, 0);
#else
  __builtin_amdgcn_tensor_load_to_lds(g0, g1, z4, z4, 0);
#endif
}
#endif  // USE_TDM

union Frag16 { v16h v; unsigned u[8]; };

// Load a 16x32 f16 fragment following the CDNA5 16-bit A-matrix layout:
// lane L holds row (row0 + L&15); K pairs (2 halfs) per dword:
//   u[0..3]: K = k0 + 8*(L>>4) + 2v ; u[4..7]: K = k0 + 16 + 8*(L>>4) + 2v
// Each half-fragment is 16B contiguous -> compiler merges into b128 loads.
// B fragments use the same loader on a [N][K] (transposed) image.
__device__ inline v16h load_frag(const h16* __restrict__ base, int ld,
                                 int row0, int k0) {
  const int lane = threadIdx.x & 31;
  const int r    = row0 + (lane & 15);
  const int hs   = lane >> 4;
  const h16* p   = base + (long)r * ld + k0 + 8 * hs;
  Frag16 f;
#pragma unroll
  for (int v = 0; v < 4; ++v) f.u[v]     = *(const unsigned*)(p + 2 * v);
#pragma unroll
  for (int v = 0; v < 4; ++v) f.u[4 + v] = *(const unsigned*)(p + 16 + 2 * v);
  return f.v;
}

__device__ inline v8f wmma16(v16h a, v16h b, v8f c) {
  return __builtin_amdgcn_wmma_f32_16x16x32_f16(false, a, false, b,
                                                (short)0, c, false, false);
}

// ---------------------------------------------------------------------------
// Fused gconv: OUT[b] = adj @ (X[b] @ W) (+bias for fp32 out modes)
//   FIN_PAD : K padding (mult of 32).  FOUT_TILES : N tiles of 16.
//   INF32   : input is fp32 [NN][INF] per sample (first layer only)
//   OUTMODE : 0 = f16 tensor out [NN][rstride]
//             1 = fp32 out + bias,  2 = fp32 out + bias + sigmoid
// ---------------------------------------------------------------------------
template <int FIN_PAD, int FOUT_TILES, bool INF32, int OUTMODE>
__global__ __launch_bounds__(256) void gconv_fused(
    const void* __restrict__ inp, const h16* __restrict__ Wt,
    const h16* __restrict__ adjh, const float* __restrict__ bias,
    void* __restrict__ outp, int out_valid, long out_sstride, int out_rstride) {
  constexpr int FOP = FOUT_TILES * 16;
  __shared__ __align__(16) h16 sX[64 * FIN_PAD];
  __shared__ __align__(16) h16 sUT[FOP * 64];   // stage-1 result, transposed [n][m]
  __shared__ __align__(16) h16 sAdj[64 * 64];

  const int tid = threadIdx.x, wave = tid >> 5, lane = tid & 31;
  const int b = blockIdx.x;
#ifdef USE_ASYNC_LDS
  bool used_async = false;
#endif

  // stage adj (pre-padded 64x64 f16) : 512 x 16B chunks
  {
    const int4* s = (const int4*)adjh;
    int4*       d = (int4*)sAdj;
#ifdef USE_ASYNC_LDS
    for (int i = tid; i < (64 * 64 * 2) / 16; i += 256)
      __builtin_amdgcn_global_load_async_to_lds_b128((gv4i*)(s + i),
                                                     (lv4i*)(d + i), 0, 0);
    used_async = true;
#else
    for (int i = tid; i < (64 * 64 * 2) / 16; i += 256) d[i] = s[i];
#endif
  }
  // stage X
  if constexpr (INF32) {
    const float* x = (const float*)inp + (long)b * (NN * INF);
    for (int i = tid; i < 64 * FIN_PAD; i += 256) {
      int r = i / FIN_PAD, c = i % FIN_PAD;
      sX[i] = (h16)((r < NN && c < INF) ? x[r * INF + c] : 0.f);
    }
  } else {
    const h16* xa = (const h16*)inp + (long)b * (NN * FIN_PAD);
    if (b + 1 < NB)
      __builtin_prefetch((const char*)inp + (long)(b + 1) * NN * FIN_PAD * 2, 0, 1);
    constexpr int CH  = 64 * FIN_PAD / 8;  // 8-half (16B) chunks
    constexpr int CHV = NN * FIN_PAD / 8;
    int4*       d = (int4*)sX;
    const int4* s = (const int4*)xa;
#if defined(USE_TDM)
    // zero the K-pad rows 57..63 with plain LDS stores
    for (int i = CHV + tid; i < CH; i += 256) d[i] = make_int4(0, 0, 0, 0);
    if (wave == 0) {             // wave-uniform: exactly one DMA issue
      tdm_load_2d(sX, xa, NN, FIN_PAD);
      tensor_wait0();            // TENSORcnt -> 0 before barrier
    }
#elif defined(USE_ASYNC_LDS)
    for (int i = tid; i < CH; i += 256) {
      if (i < CHV)
        __builtin_amdgcn_global_load_async_to_lds_b128((gv4i*)(s + i),
                                                       (lv4i*)(d + i), 0, 0);
      else
        d[i] = make_int4(0, 0, 0, 0);
    }
    used_async = true;
#else
    for (int i = tid; i < CH; i += 256)
      d[i] = (i < CHV) ? s[i] : make_int4(0, 0, 0, 0);
#endif
  }
#ifdef USE_ASYNC_LDS
  if (used_async) async_wait0();   // drain this wave's ASYNCcnt before barrier
#endif
  __syncthreads();

  // ---- stage 1 : U = X @ W  (tiles striped over 8 waves) ----
  for (int t = wave; t < 4 * FOUT_TILES; t += 8) {
    const int rt = t & 3, ct = t >> 2;
    v8f acc = {0, 0, 0, 0, 0, 0, 0, 0};
#pragma unroll
    for (int kc = 0; kc < FIN_PAD / 32; ++kc) {
      v16h a = load_frag(sX, FIN_PAD, rt * 16, kc * 32);
      v16h w = load_frag(Wt, FIN_PAD, ct * 16, kc * 32);  // W^T [N][K]
      acc = wmma16(a, w, acc);
    }
    const int n = ct * 16 + (lane & 15), hs = lane >> 4;
    __align__(16) h16 tmp[8];
#pragma unroll
    for (int v = 0; v < 8; ++v) tmp[v] = (h16)acc[v];
    *(int4*)&sUT[n * 64 + rt * 16 + 8 * hs] = *(const int4*)tmp;
  }
  __syncthreads();

  // ---- stage 2 : Z = adj @ U ----
  for (int t = wave; t < 4 * FOUT_TILES; t += 8) {
    const int rt = t & 3, ct = t >> 2;
    v8f acc = {0, 0, 0, 0, 0, 0, 0, 0};
#pragma unroll
    for (int kc = 0; kc < 2; ++kc) {
      v16h a = load_frag(sAdj, 64, rt * 16, kc * 32);
      v16h u = load_frag(sUT, 64, ct * 16, kc * 32);  // U^T [N][K=m]
      acc = wmma16(a, u, acc);
    }
    const int n = ct * 16 + (lane & 15), hs = lane >> 4;
    if constexpr (OUTMODE == 0) {
      h16* o = (h16*)outp + (long)b * out_sstride;
#pragma unroll
      for (int v = 0; v < 8; ++v) {
        int m = rt * 16 + 8 * hs + v;
        if (m < NN) o[m * out_rstride + n] = (h16)acc[v];
      }
    } else {
      float* o = (float*)outp + (long)b * out_sstride;
      if (n < out_valid) {
        float bb = bias ? bias[n] : 0.f;
#pragma unroll
        for (int v = 0; v < 8; ++v) {
          int m = rt * 16 + 8 * hs + v;
          if (m < NN) {
            float val = acc[v] + bb;
            if constexpr (OUTMODE == 2) val = 1.f / (1.f + __expf(-val));
            o[m * out_rstride + n] = val;
          }
        }
      }
    }
  }
}

// ---------------------------------------------------------------------------
// BatchNorm statistics: one feature per block; produces scale/shift so the
// apply pass is a single FMA.  (gconv bias cancels under BN -> omitted.)
// ---------------------------------------------------------------------------
__global__ __launch_bounds__(256) void bn_stats(
    const h16* __restrict__ T, int cs, int fv, const float* __restrict__ g,
    const float* __restrict__ bb, float* __restrict__ scale,
    float* __restrict__ shift) {
  const int feat = blockIdx.x;
  const int m = feat / fv, c = feat % fv;
  const h16* p = T + (long)m * cs + c;
  float s = 0.f, s2 = 0.f;
  for (int b = threadIdx.x; b < NB; b += 256) {
    float v = (float)p[(long)b * NN * cs];
    s += v; s2 += v * v;
  }
  __shared__ float ss[256], sq[256];
  ss[threadIdx.x] = s; sq[threadIdx.x] = s2;
  __syncthreads();
  for (int st = 128; st; st >>= 1) {
    if (threadIdx.x < st) {
      ss[threadIdx.x] += ss[threadIdx.x + st];
      sq[threadIdx.x] += sq[threadIdx.x + st];
    }
    __syncthreads();
  }
  if (threadIdx.x == 0) {
    float mu  = ss[0] * (1.f / NB);
    float var = sq[0] * (1.f / NB) - mu * mu;
    float sc  = g[feat] * rsqrtf(var + BN_EPS);
    scale[feat] = sc;
    shift[feat] = bb[feat] - mu * sc;
  }
}

// out = relu(scale*T + shift) (+ resid), f16, zero-fills K-pad columns.
__global__ __launch_bounds__(256) void apply_bn(
    const h16* __restrict__ T, int cs_t, const float* __restrict__ scale,
    const float* __restrict__ shift, int fv, const h16* __restrict__ resid,
    int cs_r, h16* __restrict__ out, int cs_o, int total) {
  int idx = blockIdx.x * 256 + threadIdx.x;
  if (idx >= total) return;
  const int c = idx % cs_o;
  const int rm = idx / cs_o;
  const int m = rm % NN;
  const long b = rm / NN;
  h16 o = (h16)0.f;
  if (c < fv) {
    const int feat = m * fv + c;
    float v = scale[feat] * (float)T[b * NN * cs_t + m * cs_t + c] + shift[feat];
    v = v > 0.f ? v : 0.f;
    if (resid) v += (float)resid[b * NN * cs_r + m * cs_r + c];
    o = (h16)v;
  }
  out[idx] = o;
}

// new_feat[b,n] = lab_W[label[b], n] + lab_b[n]
__global__ __launch_bounds__(256) void label_feat(
    const int* __restrict__ labels, const float* __restrict__ labW,
    const float* __restrict__ labB, h16* __restrict__ nf) {
  int idx = blockIdx.x * 256 + threadIdx.x;
  if (idx >= NB * NN) return;
  int b = idx / NN, n = idx % NN;
  nf[idx] = (h16)(labW[labels[b] * NN + n] + labB[n]);
}

// ywl[b,m,0]=nf ; [1..128]=A2 ; pad 0   (stride 160)
__global__ __launch_bounds__(256) void pack_ywl(
    const h16* __restrict__ nf, const h16* __restrict__ A2,
    h16* __restrict__ ywl, int total) {
  int idx = blockIdx.x * 256 + threadIdx.x;
  if (idx >= total) return;
  int c = idx % 160;
  int rm = idx / 160;
  int m = rm % NN;
  long b = rm / NN;
  h16 v = (h16)0.f;
  if (c == 0)            v = nf[b * NN + m];
  else if (c <= HID)     v = A2[b * NN * HID + m * HID + (c - 1)];
  ywl[idx] = v;
}

// class features: yc[b, m*25+c] = relu(bn(T5[b,m,c])), flat [B][1440] f16.
__global__ __launch_bounds__(256) void class_feat(
    const h16* __restrict__ T5, const float* __restrict__ scale,
    const float* __restrict__ shift, h16* __restrict__ yc, int total) {
  int idx = blockIdx.x * 256 + threadIdx.x;
  if (idx >= total) return;
  int j = idx % 1440;
  long b = idx / 1440;
  h16 o = (h16)0.f;
  if (j < NN * INF) {
    int m = j / INF, c = j % INF;
    float v = scale[j] * (float)T5[b * NN * 32 + m * 32 + c] + shift[j];
    o = (h16)(v > 0.f ? v : 0.f);
  }
  yc[idx] = o;
}

// logits = yc @ lin_W + lin_b  (M=8192, K=1440, N=16 pad) via WMMA
__global__ __launch_bounds__(256) void linear_wmma(
    const h16* __restrict__ yc, const h16* __restrict__ Wl,
    const float* __restrict__ linB, float* __restrict__ logits) {
  const int wave = threadIdx.x >> 5, lane = threadIdx.x & 31;
  const long row0 = (long)blockIdx.x * 64;
  if (wave < 4) {
    v8f acc = {0, 0, 0, 0, 0, 0, 0, 0};
    const h16* abase = yc + (row0 + wave * 16) * 1440;
    for (int kc = 0; kc < 45; ++kc) {
      v16h a = load_frag(abase, 1440, 0, kc * 32);
      v16h w = load_frag(Wl, 1440, 0, kc * 32);
      acc = wmma16(a, w, acc);
    }
    const int n = lane & 15, hs = lane >> 4;
    const float bb = (n < CLS) ? linB[n] : 0.f;
#pragma unroll
    for (int v = 0; v < 8; ++v) {
      long m = row0 + wave * 16 + 8 * hs + v;
      logits[m * 16 + n] = acc[v] + bb;
    }
  }
}

__global__ __launch_bounds__(256) void log_softmax12(
    const float* __restrict__ logits, float* __restrict__ out) {
  int b = blockIdx.x * 256 + threadIdx.x;
  if (b >= NB) return;
  float x[CLS], mx = -3.4e38f;
#pragma unroll
  for (int j = 0; j < CLS; ++j) { x[j] = logits[b * 16 + j]; mx = fmaxf(mx, x[j]); }
  float s = 0.f;
#pragma unroll
  for (int j = 0; j < CLS; ++j) s += __expf(x[j] - mx);
  float lse = mx + __logf(s);
#pragma unroll
  for (int j = 0; j < CLS; ++j) out[b * CLS + j] = x[j] - lse;
}

// W [Kv][Nv] fp32  ->  dst [NProws][KP] f16 transposed + zero-padded
__global__ __launch_bounds__(256) void prep_w(
    const float* __restrict__ W, h16* __restrict__ dst, int Kv, int Nv,
    int KP, int NProws) {
  int idx = blockIdx.x * 256 + threadIdx.x;
  if (idx >= NProws * KP) return;
  int n = idx / KP, k = idx % KP;
  dst[idx] = (h16)((k < Kv && n < Nv) ? W[k * Nv + n] : 0.f);
}

// adj [57][57] fp32 -> [64][64] f16 zero-padded
__global__ __launch_bounds__(256) void prep_adj(const float* __restrict__ a,
                                                h16* __restrict__ dst) {
  int idx = blockIdx.x * 256 + threadIdx.x;
  if (idx >= 64 * 64) return;
  int r = idx >> 6, c = idx & 63;
  dst[idx] = (h16)((r < NN && c < NN) ? a[r * NN + c] : 0.f);
}

// ---------------------------------------------------------------------------
// host-side orchestration
// ---------------------------------------------------------------------------
enum {
  I_X = 0, I_LAB,
  I_GCIN_W, I_GCIN_ADJ, I_GCIN_B, I_BN1_G, I_BN1_B,
  I_BS_GC1_W, I_BS_GC1_ADJ, I_BS_GC1_B, I_BS_BN1_G, I_BS_BN1_B,
  I_BS_GC2_W, I_BS_GC2_ADJ, I_BS_GC2_B, I_BS_BN2_G, I_BS_BN2_B,
  I_B1_GC1_W, I_B1_GC1_ADJ, I_B1_GC1_B, I_B1_BN1_G, I_B1_BN1_B,
  I_B1_GC2_W, I_B1_GC2_ADJ, I_B1_GC2_B, I_B1_BN2_G, I_B1_BN2_B,
  I_CORR_W, I_CORR_ADJ, I_CORR_B,
  I_ATT_W, I_ATT_ADJ, I_ATT_B,
  I_CLS_W, I_CLS_ADJ, I_CLS_B,
  I_BN2_G, I_BN2_B, I_LIN_W, I_LIN_B, I_LAB_W, I_LAB_B
};

static inline size_t aln(size_t x) { return (x + 255) & ~(size_t)255; }

extern "C" void kernel_launch(void* const* d_in, const int* in_sizes, int n_in,
                              void* d_out, int out_size, void* d_ws,
                              size_t ws_size, hipStream_t stream) {
  (void)in_sizes; (void)n_in; (void)out_size; (void)ws_size;
  char* p = (char*)d_ws;
  auto carveH = [&](size_t n) { h16* r = (h16*)p; p += aln(n * sizeof(h16)); return r; };
  auto carveF = [&](size_t n) { float* r = (float*)p; p += aln(n * sizeof(float)); return r; };

  // weights / adj images (f16, transposed + padded)
  h16* W_in  = carveH(128 * 32);
  h16* W_g1  = carveH(128 * 128);
  h16* W_g2  = carveH(128 * 128);
  h16* W_h1  = carveH(144 * 160);
  h16* W_h2  = carveH(144 * 160);
  h16* W_co  = carveH(32 * 160);
  h16* W_at  = carveH(16 * 160);
  h16* W_cl  = carveH(32 * 128);
  h16* W_li  = carveH(16 * 1440);
  h16* ADJ   = carveH(8 * 64 * 64);
  // activations
  h16* A0  = carveH((size_t)NB * NN * 128);
  h16* A1  = carveH((size_t)NB * NN * 128);   // later reused as A2
  h16* T   = carveH((size_t)NB * NN * 160);   // pre-BN scratch (all stages)
  h16* YWL = carveH((size_t)NB * NN * 160);
  h16* P3  = carveH((size_t)NB * NN * 160);   // A3 then A4
  h16* YC  = carveH((size_t)NB * 1440);
  h16* NF  = carveH((size_t)NB * NN);
  float* SC  = carveF(NN * (HID + 1));
  float* SH  = carveF(NN * (HID + 1));
  float* LG  = carveF((size_t)NB * 16);

  const float* x   = (const float*)d_in[I_X];
  const int*   lab = (const int*)d_in[I_LAB];
  float* out_corr  = (float*)d_out;
  float* out_att   = (float*)d_out + (size_t)NB * NN * INF;
  float* out_cls   = out_att + (size_t)NB * NN;

  auto F = [&](int i) { return (const float*)d_in[i]; };
  auto g1 = [](int n) { return dim3((n + 255) / 256); };

  // ---- weight prep ----
  prep_w<<<g1(128 * 32), 256, 0, stream>>>(F(I_GCIN_W), W_in, 25, 128, 32, 128);
  prep_w<<<g1(128 * 128), 256, 0, stream>>>(F(I_BS_GC1_W), W_g1, 128, 128, 128, 128);
  prep_w<<<g1(128 * 128), 256, 0, stream>>>(F(I_BS_GC2_W), W_g2, 128, 128, 128, 128);
  prep_w<<<g1(144 * 160), 256, 0, stream>>>(F(I_B1_GC1_W), W_h1, 129, 129, 160, 144);
  prep_w<<<g1(144 * 160), 256, 0, stream>>>(F(I_B1_GC2_W), W_h2, 129, 129, 160, 144);
  prep_w<<<g1(32 * 160), 256, 0, stream>>>(F(I_CORR_W), W_co, 129, 25, 160, 32);
  prep_w<<<g1(16 * 160), 256, 0, stream>>>(F(I_ATT_W), W_at, 129, 1, 160, 16);
  prep_w<<<g1(32 * 128), 256, 0, stream>>>(F(I_CLS_W), W_cl, 128, 25, 128, 32);
  prep_w<<<g1(16 * 1440), 256, 0, stream>>>(F(I_LIN_W), W_li, 1425, 12, 1440, 16);
  const int adj_src[8] = {I_GCIN_ADJ, I_BS_GC1_ADJ, I_BS_GC2_ADJ, I_B1_GC1_ADJ,
                          I_B1_GC2_ADJ, I_CORR_ADJ, I_ATT_ADJ, I_CLS_ADJ};
  for (int i = 0; i < 8; ++i)
    prep_adj<<<g1(64 * 64), 256, 0, stream>>>(F(adj_src[i]), ADJ + i * 4096);

  const int tot128 = NB * NN * 128, tot160 = NB * NN * 160;

  // ---- gcin -> BN1 -> relu -> A0 ----
  gconv_fused<32, 8, true, 0><<<NB, 256, 0, stream>>>(
      x, W_in, ADJ + 0 * 4096, nullptr, T, 0, (long)NN * 128, 128);
  bn_stats<<<NN * HID, 256, 0, stream>>>(T, 128, 128, F(I_BN1_G), F(I_BN1_B), SC, SH);
  apply_bn<<<g1(tot128), 256, 0, stream>>>(T, 128, SC, SH, 128, nullptr, 0, A0, 128, tot128);

  // ---- gcbs ----
  gconv_fused<128, 8, false, 0><<<NB, 256, 0, stream>>>(
      A0, W_g1, ADJ + 1 * 4096, nullptr, T, 0, (long)NN * 128, 128);
  bn_stats<<<NN * HID, 256, 0, stream>>>(T, 128, 128, F(I_BS_BN1_G), F(I_BS_BN1_B), SC, SH);
  apply_bn<<<g1(tot128), 256, 0, stream>>>(T, 128, SC, SH, 128, nullptr, 0, A1, 128, tot128);
  gconv_fused<128, 8, false, 0><<<NB, 256, 0, stream>>>(
      A1, W_g2, ADJ + 2 * 4096, nullptr, T, 0, (long)NN * 128, 128);
  bn_stats<<<NN * HID, 256, 0, stream>>>(T, 128, 128, F(I_BS_BN2_G), F(I_BS_BN2_B), SC, SH);
  apply_bn<<<g1(tot128), 256, 0, stream>>>(T, 128, SC, SH, 128, A0, 128, A1, 128, tot128);
  // A1 now holds A2 = gcbs output

  // ---- label concat ----
  label_feat<<<g1(NB * NN), 256, 0, stream>>>(lab, F(I_LAB_W), F(I_LAB_B), NF);
  pack_ywl<<<g1(tot160), 256, 0, stream>>>(NF, A1, YWL, tot160);

  // ---- gcbs1 (F=129, K padded 160, 9 N-tiles) ----
  gconv_fused<160, 9, false, 0><<<NB, 256, 0, stream>>>(
      YWL, W_h1, ADJ + 3 * 4096, nullptr, T, 0, (long)NN * 160, 160);
  bn_stats<<<NN * 129, 256, 0, stream>>>(T, 160, 129, F(I_B1_BN1_G), F(I_B1_BN1_B), SC, SH);
  apply_bn<<<g1(tot160), 256, 0, stream>>>(T, 160, SC, SH, 129, nullptr, 0, P3, 160, tot160);
  gconv_fused<160, 9, false, 0><<<NB, 256, 0, stream>>>(
      P3, W_h2, ADJ + 4 * 4096, nullptr, T, 0, (long)NN * 160, 160);
  bn_stats<<<NN * 129, 256, 0, stream>>>(T, 160, 129, F(I_B1_BN2_G), F(I_B1_BN2_B), SC, SH);
  apply_bn<<<g1(tot160), 256, 0, stream>>>(T, 160, SC, SH, 129, YWL, 160, P3, 160, tot160);
  // P3 now holds A4 = gcbs1 output

  // ---- heads on A4: corr (fp32+bias) and att (fp32+bias+sigmoid) ----
  gconv_fused<160, 2, false, 1><<<NB, 256, 0, stream>>>(
      P3, W_co, ADJ + 5 * 4096, F(I_CORR_B), out_corr, 25, (long)NN * 25, 25);
  gconv_fused<160, 1, false, 2><<<NB, 256, 0, stream>>>(
      P3, W_at, ADJ + 6 * 4096, F(I_ATT_B), out_att, 1, (long)NN, 1);

  // ---- class head on A2: gcout_class -> BN2 -> relu -> linear -> lsm ----
  gconv_fused<128, 2, false, 0><<<NB, 256, 0, stream>>>(
      A1, W_cl, ADJ + 7 * 4096, nullptr, T, 0, (long)NN * 32, 32);
  bn_stats<<<NN * INF, 256, 0, stream>>>(T, 32, 25, F(I_BN2_G), F(I_BN2_B), SC, SH);
  class_feat<<<g1(NB * 1440), 256, 0, stream>>>(T, SC, SH, YC, NB * 1440);
  linear_wmma<<<NB / 64, 256, 0, stream>>>(YC, W_li, F(I_LIN_B), LG);
  log_softmax12<<<g1(NB), 256, 0, stream>>>(LG, out_cls);
}
